// WindspeedLSTMHelper_3281355014162
// MI455X (gfx1250) — compile-verified, matrix operand order checked by reading, not run on hardware
//
#include <hip/hip_runtime.h>

typedef __attribute__((ext_vector_type(16))) _Float16 v16h;
typedef __attribute__((ext_vector_type(8)))  _Float16 v8h;
typedef __attribute__((ext_vector_type(8)))  float    v8f;

constexpr int kSeq   = 512;
constexpr int kBatch = 1024;
constexpr int kH     = 64;    // hidden == input
constexpr int kG     = 256;   // 4*H gate width
constexpr int kK     = 128;   // concat [x | h] depth
constexpr int kBM    = 16;    // batch rows per workgroup (one WMMA M-tile)
constexpr int kThreads = 512; // 16 wave32 waves: one N-tile per wave
constexpr int kElems = (kBM * kH) / kThreads;   // 2 state elements per thread

// LDS row strides (padded to break 64-bank alignment)
constexpr int kWS = kK + 8;   // half stride for weight rows  (272 B)
constexpr int kAS = kK + 8;   // half stride for A rows       (272 B)
constexpr int kGS = kG + 4;   // float stride for gate rows   (1040 B)

// Branchless tanh: hardware v_tanh_f32 when the builtin exists, otherwise a
// saturation-safe exp/rcp form. No EXEC manipulation either way.
__device__ __forceinline__ float fast_tanhf(float x) {
#if __has_builtin(__builtin_amdgcn_tanhf)
    return __builtin_amdgcn_tanhf(x);
#else
    const float t = __expf(-2.0f * __builtin_fabsf(x));       // t in (0,1]
    const float r = (1.0f - t) * __builtin_amdgcn_rcpf(1.0f + t);
    return __builtin_copysignf(r, x);
#endif
}

__device__ __forceinline__ float fast_sigmoidf(float x) {
    return 0.5f + 0.5f * fast_tanhf(0.5f * x);
}

__device__ __forceinline__ v16h join8(v8h lo, v8h hi) {
    return __builtin_shufflevector(lo, hi,
        0, 1, 2, 3, 4, 5, 6, 7, 8, 9, 10, 11, 12, 13, 14, 15);
}

__global__ __launch_bounds__(kThreads)
void windspeed_lstm_persistent(const float* __restrict__ x,
                               const float* __restrict__ W_ih,
                               const float* __restrict__ W_hh,
                               const float* __restrict__ b_ih,
                               const float* __restrict__ b_hh,
                               float* __restrict__ out)
{
    // Wcat[n][k]: k<64 -> W_ih[n][k], k>=64 -> W_hh[n][k-64], f16.
    __shared__ _Float16 Wlds[kG * kWS];          // ~68 KB (staging only)
    __shared__ _Float16 Alds[kBM * kAS];         // [x_t | h_{t-1}] tile, f16
    __shared__ float    Glds[kBM * kGS];         // biased gates, f32

    const int tid  = threadIdx.x;
    const int lane = tid & 31;
    const int wave = tid >> 5;                   // 16 waves
    const int bb   = blockIdx.x * kBM;           // batch base row

    // ---- one-time: weights into LDS (f32 -> f16); 512 threads = half-row each
    {
        const int n  = tid >> 1;                 // gate row 0..255
        const int kh = (tid & 1) * (kH / 2);     // k half [0,32) or [32,64)
        for (int k = kh; k < kh + kH / 2; ++k) {
            Wlds[n * kWS + k]      = (_Float16)W_ih[n * kH + k];
            Wlds[n * kWS + kH + k] = (_Float16)W_hh[n * kH + k];
        }
    }

    // ---- init: c=0 (registers), h-part of A = 0, load x[0] ----
    float c_reg[kElems];
    float x_reg[kElems];
    #pragma unroll
    for (int e = 0; e < kElems; ++e) {
        const int idx = e * kThreads + tid;      // 0..1023 over (m,j)
        const int m = idx >> 6;
        const int j = idx & 63;
        c_reg[e] = 0.0f;
        Alds[m * kAS + kH + j] = (_Float16)0.0f;
        Alds[m * kAS + j] =
            (_Float16)__builtin_nontemporal_load(&x[(size_t)bb * kH + idx]);
    }
    __syncthreads();

    // ---- per-wave WMMA fragment addressing (ISA 7.12.2, 16-bit layouts) ----
    const int row16 = lane & 15;                 // A: M row / B: N column (within tile)
    const int sel   = (lane >> 4) & 1;           // half-wave select
    const int kb0   = sel * 8;                   // K sub-base within a 32-K tile
    const int ncol  = wave * 16 + row16;         // this wave's single gate column set

    // Per-lane gate bias for this lane's column; folded into the WMMA
    // accumulator init (C/D layout: a lane owns one column for all 8 rows).
    const float bias = b_ih[ncol] + b_hh[ncol];
    const v8f biasv = {bias, bias, bias, bias, bias, bias, bias, bias};

    // ---- hoist loop-invariant B (weight) fragments into registers ----
    // 1 N-tile x 4 K-tiles x 8 VGPRs = 32 VGPRs/lane, resident for all 512 steps.
    v16h bf[4];
    #pragma unroll
    for (int kt = 0; kt < 4; ++kt) {
        const int kb = kt * 32 + kb0;
        bf[kt] = join8(*(const v8h*)&Wlds[ncol * kWS + kb],
                       *(const v8h*)&Wlds[ncol * kWS + kb + 16]);
    }

    for (int t = 0; t < kSeq; ++t) {
        // Prefetch next timestep's x tile into registers; overlaps the WMMAs.
        const int t1 = (t + 1 < kSeq) ? (t + 1) : (kSeq - 1);
        const size_t xb = ((size_t)t1 * kBatch + bb) * kH;
        #pragma unroll
        for (int e = 0; e < kElems; ++e)
            x_reg[e] = __builtin_nontemporal_load(&x[xb + e * kThreads + tid]);

        // gates(16x256) = A(16x128) * Wcat^T(128x256) + bias, f32 accumulate.
        // Two independent 2-deep WMMA chains; bias rides in chain 'a' init.
        v8f acc_a = biasv, acc_b = {};
        v16h af[4];
        #pragma unroll
        for (int kt = 0; kt < 4; ++kt) {
            const int kb = kt * 32 + kb0;
            af[kt] = join8(*(const v8h*)&Alds[row16 * kAS + kb],
                           *(const v8h*)&Alds[row16 * kAS + kb + 16]);
        }
        acc_a = __builtin_amdgcn_wmma_f32_16x16x32_f16(false, af[0], false, bf[0], (short)0, acc_a, false, false);
        acc_b = __builtin_amdgcn_wmma_f32_16x16x32_f16(false, af[1], false, bf[1], (short)0, acc_b, false, false);
        acc_a = __builtin_amdgcn_wmma_f32_16x16x32_f16(false, af[2], false, bf[2], (short)0, acc_a, false, false);
        acc_b = __builtin_amdgcn_wmma_f32_16x16x32_f16(false, af[3], false, bf[3], (short)0, acc_b, false, false);
        const v8f acc = acc_a + acc_b;

        // Scatter accumulator to gate LDS (D layout: VGPR r -> M = r + 8*sel)
        const int mb = sel * 8;
        #pragma unroll
        for (int r = 0; r < 8; ++r) {
            Glds[(mb + r) * kGS + ncol] = acc[r];
        }
        __syncthreads();

        // ---- elementwise LSTM cell: i,f,g,o -> c,h (branchless TRANS ops) ----
        const size_t ob = ((size_t)t * kBatch + bb) * kH;
        #pragma unroll
        for (int e = 0; e < kElems; ++e) {
            const int idx = e * kThreads + tid;
            const int m = idx >> 6;
            const int j = idx & 63;
            const float iv = fast_sigmoidf(Glds[m * kGS + j]);
            const float fv = fast_sigmoidf(Glds[m * kGS + kH + j]);
            const float gv = fast_tanhf  (Glds[m * kGS + 2 * kH + j]);
            const float ov = fast_sigmoidf(Glds[m * kGS + 3 * kH + j]);
            const float c  = fv * c_reg[e] + iv * gv;
            c_reg[e] = c;
            const float h = ov * fast_tanhf(c);
            __builtin_nontemporal_store(h, &out[ob + idx]); // streamed, write-once
            Alds[m * kAS + kH + j] = (_Float16)h;           // recurrent input for t+1
            Alds[m * kAS + j]      = (_Float16)x_reg[e];    // x for t+1
        }
        __syncthreads();
    }
}

extern "C" void kernel_launch(void* const* d_in, const int* in_sizes, int n_in,
                              void* d_out, int out_size, void* d_ws, size_t ws_size,
                              hipStream_t stream) {
    (void)in_sizes; (void)n_in; (void)d_ws; (void)ws_size; (void)out_size;
    const float* x    = (const float*)d_in[0];
    const float* W_ih = (const float*)d_in[1];
    const float* W_hh = (const float*)d_in[2];
    const float* b_ih = (const float*)d_in[3];
    const float* b_hh = (const float*)d_in[4];
    float* out = (float*)d_out;

    dim3 grid(kBatch / kBM);   // 64 independent batch tiles
    dim3 block(kThreads);      // 16 wave32 waves, one gate N-tile each
    windspeed_lstm_persistent<<<grid, block, 0, stream>>>(
        x, W_ih, W_hh, b_ih, b_hh, out);
}
